// GroupedQueryAttention_29841432772809
// MI455X (gfx1250) — matrix operand order repaired and checked by reading, not compiled, not run on hardware
//
#include <hip/hip_runtime.h>
#include <hip/hip_bf16.h>

typedef __bf16 bf16;
typedef __attribute__((ext_vector_type(16))) __bf16 v16bf;
typedef __attribute__((ext_vector_type(8)))  __bf16 v8bf;
typedef __attribute__((ext_vector_type(4)))  __bf16 v4bf;
typedef __attribute__((ext_vector_type(8)))  float  v8f;

static constexpr int Bv = 2, Tv = 2048, Dv = 4096, Hv = 32, KVv = 8, HDv = 128;
static constexpr int HHD  = Hv * HDv;    // 4096
static constexpr int KVHD = KVv * HDv;   // 1024
static constexpr int NREP = Hv / KVv;    // 4

// ---------------------------------------------------------------------------
// WMMA helpers (CDNA5 16x16x32 bf16, f32 accumulate)
// ---------------------------------------------------------------------------
__device__ __forceinline__ v8f wmma_bf16(v16bf a, v16bf b, v8f c) {
  // 8 args: (neg_a, A, neg_b, B, c_mod, C, reuse_a, reuse_b)
  return __builtin_amdgcn_wmma_f32_16x16x32_bf16(false, a, false, b, (short)0, c,
                                                 false, false);
}

// A-matrix 16x32 (MxK), row-major source with leading dim `ld` (elements).
// Contiguous-k layout used here (not ISA table order): lanes 0-15 = rows 0-15, K=0..15;
//             lanes 16-31 = rows 0-15, K=16..31.
__device__ __forceinline__ v16bf load_a_frag(const bf16* base, int ld) {
  const int lane = threadIdx.x & 31;
  const bf16* p = base + (size_t)(lane & 15) * ld + ((lane < 16) ? 0 : 16);
  union { v16bf v; v8bf h[2]; } u;
  u.h[0] = *(const v8bf*)(p);
  u.h[1] = *(const v8bf*)(p + 8);
  return u.v;
}

// B-matrix 32x16 (KxN) loaded from row-major [N][K] memory with leading dim ld.
// Contiguous-k layout used here (same as load_a_frag): lanes 0-15 = col n, K=0..15 ; lanes 16-31 = col n, K=16..31.
__device__ __forceinline__ v16bf load_b_frag(const bf16* base, int ld) {
  const int lane = threadIdx.x & 31;
  const bf16* p = base + (size_t)(lane & 15) * ld + ((lane < 16) ? 0 : 16);
  return *(const v16bf*)(p);
}

// ---------------------------------------------------------------------------
// fp32 -> bf16 conversion, 128-bit loads / 64-bit stores (n divisible by 4)
// ---------------------------------------------------------------------------
__global__ void cvt_kernel(const float* __restrict__ in, bf16* __restrict__ out,
                           size_t n4) {
  const float4* in4 = (const float4*)in;
  v4bf* out4 = (v4bf*)out;
  size_t stride = (size_t)gridDim.x * blockDim.x;
  for (size_t i = (size_t)blockIdx.x * blockDim.x + threadIdx.x; i < n4; i += stride) {
    float4 f = in4[i];
    v4bf h;
    h[0] = (bf16)f.x; h[1] = (bf16)f.y; h[2] = (bf16)f.z; h[3] = (bf16)f.w;
    out4[i] = h;
  }
}

// ---------------------------------------------------------------------------
// Generic GEMM: C[M,N] = A[M,K] @ W[N,K]^T   (A, W bf16 row-major)
// One wave -> 32x64 output tile (2 M-frags x 4 N-frags, B reused across M).
// 8 waves per block cover a 32x512 super-tile.
// ---------------------------------------------------------------------------
template <typename OutT>
__global__ __launch_bounds__(256) void gemm_bf16_wmma(
    const bf16* __restrict__ A, const bf16* __restrict__ W, OutT* __restrict__ C,
    int M, int N, int K) {
  const int wave = threadIdx.x >> 5;
  const int lane = threadIdx.x & 31;
  const int m0 = blockIdx.y * 32;
  const int n0 = (blockIdx.x * 8 + wave) * 64;
  if (n0 >= N || m0 >= M) return;

  v8f c[2][4] = {};
  for (int k0 = 0; k0 < K; k0 += 32) {
    v16bf a0 = load_a_frag(A + (size_t)m0 * K + k0, K);
    v16bf a1 = load_a_frag(A + (size_t)(m0 + 16) * K + k0, K);
#pragma unroll
    for (int t = 0; t < 4; ++t) {
      v16bf b = load_b_frag(W + (size_t)(n0 + t * 16) * K + k0, K);
      c[0][t] = wmma_bf16(a0, b, c[0][t]);
      c[1][t] = wmma_bf16(a1, b, c[1][t]);
    }
  }
  const int col = lane & 15;
  const int rb = (lane < 16) ? 0 : 8;
#pragma unroll
  for (int i = 0; i < 2; ++i)
#pragma unroll
    for (int t = 0; t < 4; ++t)
#pragma unroll
      for (int j = 0; j < 8; ++j)
        C[(size_t)(m0 + i * 16 + rb + j) * N + n0 + t * 16 + col] =
            (OutT)c[i][t][j];
}

// ---------------------------------------------------------------------------
// RoPE in place on bf16 [B, T, nheads, HD]; freqs fp32 [T, HD/2, 2]
// ---------------------------------------------------------------------------
__global__ void rope_kernel(bf16* __restrict__ q, const float* __restrict__ freqs,
                            int nheads) {
  const size_t n = (size_t)Bv * Tv * nheads * (HDv / 2);
  size_t stride = (size_t)gridDim.x * blockDim.x;
  for (size_t i = (size_t)blockIdx.x * blockDim.x + threadIdx.x; i < n; i += stride) {
    int p = (int)(i % (HDv / 2));
    size_t r = i / (HDv / 2);
    int h = (int)(r % nheads);
    size_t r2 = r / nheads;
    int t = (int)(r2 % Tv);
    int b = (int)(r2 / Tv);
    float c = freqs[(size_t)t * HDv + 2 * p];
    float s = freqs[(size_t)t * HDv + 2 * p + 1];
    size_t base = (((size_t)b * Tv + t) * nheads + h) * HDv + 2 * p;
    float x0 = (float)q[base], x1 = (float)q[base + 1];
    q[base]     = (bf16)(x0 * c - x1 * s);
    q[base + 1] = (bf16)(x0 * s + x1 * c);
  }
}

// ---------------------------------------------------------------------------
// V transpose: [B,T,KV,HD] -> [B,KV,HD,T]
// ---------------------------------------------------------------------------
__global__ void vtrans_kernel(const bf16* __restrict__ v, bf16* __restrict__ vt) {
  const size_t n = (size_t)Bv * Tv * KVv * HDv;
  size_t stride = (size_t)gridDim.x * blockDim.x;
  for (size_t i = (size_t)blockIdx.x * blockDim.x + threadIdx.x; i < n; i += stride) {
    int d = (int)(i % HDv);
    size_t r = i / HDv;
    int kv = (int)(r % KVv);
    size_t r2 = r / KVv;
    int t = (int)(r2 % Tv);
    int b = (int)(r2 / Tv);
    vt[(((size_t)b * KVv + kv) * HDv + d) * Tv + t] = v[i];
  }
}

// ---------------------------------------------------------------------------
// Flash attention, causal, GQA. One wave per (b, h, 16-query tile).
// Q,K: bf16 row-major [B*T, heads*HD] (post-RoPE). Vt: bf16 [B,KV,HD,T].
// ---------------------------------------------------------------------------
__global__ __launch_bounds__(32) void attn_kernel(const bf16* __restrict__ Q,
                                                  const bf16* __restrict__ K,
                                                  const bf16* __restrict__ Vt,
                                                  bf16* __restrict__ O) {
  const int lane = threadIdx.x & 31;
  const int qt = blockIdx.x, h = blockIdx.y, b = blockIdx.z;
  const int kvh = h / NREP;
  const int q0 = qt * 16;

  __shared__ bf16 pbuf[16 * 40];  // P tile 16x32, padded stride 40 halves

  const bf16* qbase = Q + ((size_t)b * Tv + q0) * HHD + (size_t)h * HDv;
  v16bf qf[4];
#pragma unroll
  for (int j = 0; j < 4; ++j) qf[j] = load_a_frag(qbase + j * 32, HHD);

  v8f o[8] = {};
  float m[8], l[8];
#pragma unroll
  for (int j = 0; j < 8; ++j) { m[j] = -3.0e38f; l[j] = 0.0f; }

  const int col = lane & 15;
  const int rb = (lane < 16) ? 0 : 8;
  const float scale = 0.08838834764831845f;  // 1/sqrt(128)

  const bf16* vbase = Vt + (((size_t)b * KVv + kvh) * HDv) * Tv;

  for (int k0 = 0; k0 < q0 + 16; k0 += 32) {
    // ---- scores: S[16x32] = Q (16x128) @ K^T ----
    v8f s0 = {}, s1 = {};
    const bf16* kb0 = K + ((size_t)b * Tv + k0) * KVHD + (size_t)kvh * HDv;
#pragma unroll
    for (int j = 0; j < 4; ++j) {
      s0 = wmma_bf16(qf[j], load_b_frag(kb0 + j * 32, KVHD), s0);
      s1 = wmma_bf16(qf[j], load_b_frag(kb0 + (size_t)16 * KVHD + j * 32, KVHD), s1);
    }

    // ---- online softmax (per-row stats across 16-lane halves) ----
    float alpha[8];
#pragma unroll
    for (int j = 0; j < 8; ++j) {
      const int qrow = q0 + rb + j;
      float a = s0[j] * scale; if (k0 + col > qrow)      a = -3.0e38f;
      float c = s1[j] * scale; if (k0 + 16 + col > qrow) c = -3.0e38f;
      float mx = fmaxf(a, c);
#pragma unroll
      for (int off = 8; off >= 1; off >>= 1) mx = fmaxf(mx, __shfl_xor(mx, off));
      const float mn = fmaxf(m[j], mx);
      alpha[j] = __expf(m[j] - mn);
      m[j] = mn;
      const float pa = __expf(a - mn);
      const float pb = __expf(c - mn);
      float rs = pa + pb;
#pragma unroll
      for (int off = 8; off >= 1; off >>= 1) rs += __shfl_xor(rs, off);
      l[j] = l[j] * alpha[j] + rs;
      pbuf[(rb + j) * 40 + col]      = (bf16)pa;
      pbuf[(rb + j) * 40 + 16 + col] = (bf16)pb;
    }
#pragma unroll
    for (int t = 0; t < 8; ++t)
#pragma unroll
      for (int j = 0; j < 8; ++j) o[t][j] *= alpha[j];

    __syncthreads();  // C-layout -> A-layout relayout via LDS
    v16bf pf = load_a_frag(pbuf, 40);

    // ---- O += P (16x32) @ V (32x128) ----
#pragma unroll
    for (int t = 0; t < 8; ++t)
      o[t] = wmma_bf16(pf, load_b_frag(vbase + (size_t)(t * 16) * Tv + k0, Tv), o[t]);
    __syncthreads();
  }

  bf16* obase = O + ((size_t)b * Tv + q0) * HHD + (size_t)h * HDv;
#pragma unroll
  for (int j = 0; j < 8; ++j) {
    const float inv = 1.0f / l[j];
#pragma unroll
    for (int t = 0; t < 8; ++t)
      obase[(size_t)(rb + j) * HHD + t * 16 + col] = (bf16)(o[t][j] * inv);
  }
}

// ---------------------------------------------------------------------------
// launcher
// ---------------------------------------------------------------------------
extern "C" void kernel_launch(void* const* d_in, const int* in_sizes, int n_in,
                              void* d_out, int out_size, void* d_ws, size_t ws_size,
                              hipStream_t stream) {
  (void)in_sizes; (void)n_in; (void)out_size; (void)ws_size;
  const float* x  = (const float*)d_in[0];
  const float* fr = (const float*)d_in[1];
  const float* Wq = (const float*)d_in[2];
  const float* Wk = (const float*)d_in[3];
  const float* Wv = (const float*)d_in[4];
  const float* Wo = (const float*)d_in[5];
  float* out = (float*)d_out;

  const size_t NX  = (size_t)Bv * Tv * Dv;   // 16777216 (also Q, O sizes)
  const size_t NWQ = (size_t)HHD * Dv;       // 16777216 (Wq, Wo)
  const size_t NWK = (size_t)KVHD * Dv;      // 4194304  (Wk, Wv, K, V)

  bf16* ws  = (bf16*)d_ws;
  bf16* xb  = ws;               // x bf16; reused as attention output Ob
  bf16* wqb = xb + NX;          // Wq bf16; reused as V^T after Q-proj
  bf16* wkb = wqb + NWQ;
  bf16* wvb = wkb + NWK;
  bf16* wob = wvb + NWK;
  bf16* qb  = wob + NWQ;
  bf16* kbb = qb + NX;
  bf16* vbb = kbb + NWK;
  bf16* vt  = wqb;              // alias: Wq dead after first GEMM
  bf16* ob  = xb;               // alias: x dead after V GEMM

  // 1) fp32 -> bf16 (vectorized x4)
  cvt_kernel<<<2048, 256, 0, stream>>>(x,  xb,  NX  / 4);
  cvt_kernel<<<2048, 256, 0, stream>>>(Wq, wqb, NWQ / 4);
  cvt_kernel<<<1024, 256, 0, stream>>>(Wk, wkb, NWK / 4);
  cvt_kernel<<<1024, 256, 0, stream>>>(Wv, wvb, NWK / 4);
  cvt_kernel<<<2048, 256, 0, stream>>>(Wo, wob, NWQ / 4);

  const int M = Bv * Tv;  // 4096
  // 2) projections (32-row M tiles)
  gemm_bf16_wmma<bf16><<<dim3(HHD / 512,  M / 32), 256, 0, stream>>>(xb, wqb, qb,  M, HHD,  Dv);
  gemm_bf16_wmma<bf16><<<dim3(KVHD / 512, M / 32), 256, 0, stream>>>(xb, wkb, kbb, M, KVHD, Dv);
  gemm_bf16_wmma<bf16><<<dim3(KVHD / 512, M / 32), 256, 0, stream>>>(xb, wvb, vbb, M, KVHD, Dv);

  // 3) RoPE on Q and K
  rope_kernel<<<4096, 256, 0, stream>>>(qb,  fr, Hv);
  rope_kernel<<<1024, 256, 0, stream>>>(kbb, fr, KVv);

  // 4) V -> V^T  (then x buffer is free for Ob)
  vtrans_kernel<<<4096, 256, 0, stream>>>(vbb, vt);

  // 5) causal flash attention
  attn_kernel<<<dim3(Tv / 16, Hv, Bv), 32, 0, stream>>>(qb, kbb, vt, ob);

  // 6) output projection, f32 result
  gemm_bf16_wmma<float><<<dim3(Dv / 512, M / 32), 256, 0, stream>>>(ob, wob, out, M, Dv, Dv);
}
